// SelfAttention_79869211836706
// MI455X (gfx1250) — matrix operand order repaired and checked by reading, not compiled, not run on hardware
//
#include <hip/hip_runtime.h>
#include <hip/hip_bf16.h>

typedef __bf16 bf16_t;
typedef __attribute__((ext_vector_type(16))) __bf16 v16bf;
typedef __attribute__((ext_vector_type(8)))  __bf16 v8bf;
typedef __attribute__((ext_vector_type(8)))  float  v8f;
typedef __attribute__((ext_vector_type(16))) float  v16f;

static constexpr int B_   = 4;
static constexpr int S_   = 4096;
static constexpr int DIN  = 768;
static constexpr int DOUT = 64;
static constexpr int ROWS = B_ * S_;   // 16384

__device__ __forceinline__ v8f wmma_bf16(v16bf a, v16bf b, v8f c) {
  // v_wmma_f32_16x16x32_bf16  (neg_a, A, neg_b, B, c_mod, C, reuse_a, reuse_b)
  return __builtin_amdgcn_wmma_f32_16x16x32_bf16(false, a, false, b, (short)0, c,
                                                 false, false);
}

// ---------------------------------------------------------------------------
// Kernel 1: QKV projection.  out = x @ W^T, per-wave 16x16 tile, K=768 loop.
// Stores: qb [ROWS,64] bf16 (pre-scaled by 1/8), kb [ROWS,64] bf16,
//         vt [B,64,S]  bf16 (transposed so PV B-fragments are contiguous).
// ---------------------------------------------------------------------------
__global__ __launch_bounds__(256) void qkv_kernel(
    const float* __restrict__ x,  const float* __restrict__ Wq,
    const float* __restrict__ Wk, const float* __restrict__ Wv,
    bf16_t* __restrict__ qb, bf16_t* __restrict__ kb, bf16_t* __restrict__ vt)
{
  const int lane = threadIdx.x & 31;
  const int widx = threadIdx.x >> 5;
  const int task = blockIdx.x * 8 + widx;      // (ROWS/16)*12 = 12288 tasks
  const int mtile = task / 12;
  const int rem   = task % 12;
  const int mat   = rem >> 2;                  // 0=Q 1=K 2=V
  const int ntile = rem & 3;

  const int half = lane >> 4;
  const int l16  = lane & 15;
  const int row  = mtile * 16 + l16;           // A-fragment row of this lane
  const int ncol = ntile * 16 + l16;           // B-fragment column of this lane

  const float* W = (mat == 0) ? Wq : (mat == 1) ? Wk : Wv;

  v8f acc = {};
  for (int k0 = 0; k0 < DIN; k0 += 32) {
    // A fragment (16x32 bf16): lane holds K = 16*half..+15
    const float* pa = x + (size_t)row * DIN + k0 + 16 * half;
    v8f f0 = *(const v8f*)pa;
    v8f f1 = *(const v8f*)(pa + 8);
    v16bf a;
#pragma unroll
    for (int i = 0; i < 8; ++i) { a[i] = (bf16_t)f0[i]; a[i + 8] = (bf16_t)f1[i]; }

    // B fragment (32x16 bf16): B[k][n] = W[n][k]; lane n holds K = 16*half..+15
    const float* pb = W + (size_t)ncol * DIN + k0 + 16 * half;
    v16f fb = *(const v16f*)pb;
    v16bf b;
#pragma unroll
    for (int i = 0; i < 16; ++i) b[i] = (bf16_t)fb[i];

    acc = wmma_bf16(a, b, acc);
  }

  // C layout: lane column = ncol, VGPR i -> row mtile*16 + 8*half + i
  if (mat == 2) {
    const int bidx = (mtile * 16) >> 12;       // /S_
    const int s0   = (mtile * 16) & (S_ - 1);
    v8bf o;
#pragma unroll
    for (int i = 0; i < 8; ++i) o[i] = (bf16_t)acc[i];
    *(v8bf*)(vt + ((size_t)(bidx * DOUT + ncol)) * S_ + s0 + 8 * half) = o;
  } else {
    bf16_t* dst = (mat == 0) ? qb : kb;
    const float sc = (mat == 0) ? 0.125f : 1.0f;   // fold 1/sqrt(64) into q
    const int r0 = mtile * 16 + 8 * half;
#pragma unroll
    for (int i = 0; i < 8; ++i)
      dst[(size_t)(r0 + i) * DOUT + ncol] = (bf16_t)(acc[i] * sc);
  }
}

// ---------------------------------------------------------------------------
// Async stage of one 32-key K/V tile into LDS (4 KB + 4 KB), 128 threads,
// 4 x global_load_async_to_lds_b128 per thread, tracked by ASYNCcnt.
// ---------------------------------------------------------------------------
__device__ __forceinline__ void stage_kv(
    const bf16_t* __restrict__ kb, const bf16_t* __restrict__ vt,
    int b, int kb0, unsigned kdst, unsigned vdst, int tid)
{
  // K tile: rows kb0..kb0+31 are contiguous: one flat 4 KB region.
  const unsigned long long kga0 =
      (unsigned long long)(size_t)kb + ((unsigned long long)(b * S_ + kb0) * DOUT) * 2;
#pragma unroll
  for (int r = 0; r < 2; ++r) {
    const int c = tid + r * 128;               // chunk 0..255, 16B each
    unsigned long long kga = kga0 + (unsigned)(c * 16);
    asm volatile("global_load_async_to_lds_b128 %0, %1, off"
                 :: "v"(kdst + c * 16), "v"(kga) : "memory");
    // V tile: 64 rows (d) x 32 keys; row d has 4 x 16B chunks.
    const int d = c >> 2, part = c & 3;
    unsigned long long vga =
        (unsigned long long)(size_t)vt +
        (((unsigned long long)(b * DOUT + d) * S_ + kb0) * 2) + (unsigned)(part * 16);
    asm volatile("global_load_async_to_lds_b128 %0, %1, off"
                 :: "v"(vdst + c * 16), "v"(vga) : "memory");
  }
}

// ---------------------------------------------------------------------------
// Kernel 2: flash attention.  4 waves/block share double-buffered K/V tiles
// staged asynchronously into LDS; each wave owns a 16-query tile.
// ---------------------------------------------------------------------------
__global__ __launch_bounds__(128) void attn_kernel(
    const bf16_t* __restrict__ qb, const bf16_t* __restrict__ kb,
    const bf16_t* __restrict__ vt, float* __restrict__ out)
{
  // [ K buf0 | K buf1 | V buf0 | V buf1 | P(4 waves) ] = 4+4+4+4+4 KB
  __shared__ __align__(64) bf16_t smem[10240];
  // NOTE: keep all LDS addressing as scalar arithmetic (no const pointer
  // arrays!) so clang does not promote addrspacecast exprs to .rodata.
  const unsigned smem_base = (unsigned)(size_t)&smem[0];  // low 32 bits = LDS offset

  const int tid  = threadIdx.x;
  const int lane = tid & 31;
  const int widx = tid >> 5;
  const int half = lane >> 4;
  const int l16  = lane & 15;

  const int qt = blockIdx.x * 4 + widx;       // 1024 query tiles, same b per block
  const int b  = qt >> 8;                     // / (S_/16)
  const int q0 = (qt & 255) * 16;

  bf16_t* const myp = smem + 8192 + widx * 512;   // 16x32 P tile, wave-private

  // Q A-fragments for both k-steps of D=64 (pre-scaled by 1/8)
  v16bf aq[2];
  {
    const bf16_t* base = qb + (size_t)(b * S_ + q0 + l16) * DOUT;
#pragma unroll
    for (int ks = 0; ks < 2; ++ks) {
      v8bf c0 = *(const v8bf*)(base + ks * 32 + 16 * half);
      v8bf c1 = *(const v8bf*)(base + ks * 32 + 16 * half + 8);
#pragma unroll
      for (int i = 0; i < 8; ++i) { aq[ks][i] = c0[i]; aq[ks][i + 8] = c1[i]; }
    }
  }

  v8f o0 = {}, o1 = {}, o2 = {}, o3 = {};
  float m[8], l[8];
#pragma unroll
  for (int i = 0; i < 8; ++i) { m[i] = -3.0e38f; l[i] = 0.0f; }

  // Prologue: stage tile 0 into buffer 0.
  stage_kv(kb, vt, b, 0, smem_base, smem_base + 8192, tid);

  for (int kb0 = 0; kb0 < S_; kb0 += 32) {
    const int buf = (kb0 >> 5) & 1;
    const int nxt = buf ^ 1;
    const bool have_next = (kb0 + 32) < S_;
    if (have_next)   // prefetch next tile into the other buffer (freed by the
                     // trailing barrier of the previous iteration)
      stage_kv(kb, vt, b, kb0 + 32,
               smem_base + nxt * 4096, smem_base + 8192 + nxt * 4096, tid);

    // Async loads complete in order: <=4 outstanding means tile `buf` is done.
    if (have_next) asm volatile("s_wait_asynccnt 0x4" ::: "memory");
    else           asm volatile("s_wait_asynccnt 0x0" ::: "memory");
    __syncthreads();

    bf16_t* const kbl = smem + buf * 2048;          // K tile: [key 0..31][d 0..63]
    bf16_t* const vbl = smem + 4096 + buf * 2048;   // V tile: [d 0..63][key 0..31]

    // ---- scores: 16 queries x 32 keys from LDS K tile ----
    v8f c0 = {}, c1 = {};
    {
      const bf16_t* kr0 = kbl + (size_t)l16 * DOUT + 16 * half;         // keys 0-15
      v16bf bk0a = *(const v16bf*)kr0;
      v16bf bk0b = *(const v16bf*)(kr0 + 32);
      c0 = wmma_bf16(aq[0], bk0a, c0);
      c0 = wmma_bf16(aq[1], bk0b, c0);
      const bf16_t* kr1 = kbl + (size_t)(16 + l16) * DOUT + 16 * half;  // keys 16-31
      v16bf bk1a = *(const v16bf*)kr1;
      v16bf bk1b = *(const v16bf*)(kr1 + 32);
      c1 = wmma_bf16(aq[0], bk1a, c1);
      c1 = wmma_bf16(aq[1], bk1b, c1);
    }

    // ---- online softmax (row = 8*half + i, spread over a 16-lane half) ----
    float rm[8], p0[8], p1[8], rs[8];
#pragma unroll
    for (int i = 0; i < 8; ++i) rm[i] = fmaxf(c0[i], c1[i]);
#pragma unroll
    for (int mk = 1; mk <= 8; mk <<= 1)
#pragma unroll
      for (int i = 0; i < 8; ++i) rm[i] = fmaxf(rm[i], __shfl_xor(rm[i], mk, 32));

#pragma unroll
    for (int i = 0; i < 8; ++i) {
      float nm = fmaxf(m[i], rm[i]);
      float al = __expf(m[i] - nm);
      m[i] = nm;
      p0[i] = __expf(c0[i] - nm);
      p1[i] = __expf(c1[i] - nm);
      rs[i] = p0[i] + p1[i];
      l[i] *= al;
      o0[i] *= al; o1[i] *= al; o2[i] *= al; o3[i] *= al;
    }
#pragma unroll
    for (int mk = 1; mk <= 8; mk <<= 1)
#pragma unroll
      for (int i = 0; i < 8; ++i) rs[i] += __shfl_xor(rs[i], mk, 32);
#pragma unroll
    for (int i = 0; i < 8; ++i) l[i] += rs[i];

    // ---- P: C-layout -> LDS [row][key] -> A-fragment (wave-private region;
    //      per-wave DS ops are in-order, no barrier needed) ----
#pragma unroll
    for (int i = 0; i < 8; ++i) {
      myp[(8 * half + i) * 32 + l16]      = (bf16_t)p0[i];
      myp[(8 * half + i) * 32 + 16 + l16] = (bf16_t)p1[i];
    }
    v16bf ap;
    {
      const bf16_t* pr = myp + l16 * 32;
      v8bf a0 = *(const v8bf*)(pr + 16 * half);
      v8bf a1 = *(const v8bf*)(pr + 16 * half + 8);
#pragma unroll
      for (int i = 0; i < 8; ++i) { ap[i] = a0[i]; ap[i + 8] = a1[i]; }
    }

    // ---- P @ V from LDS V tile: 4 output n-tiles, K=32 keys each ----
#pragma unroll
    for (int nt = 0; nt < 4; ++nt) {
      const bf16_t* vr = vbl + (size_t)(nt * 16 + l16) * 32 + 16 * half;
      v16bf bv = *(const v16bf*)vr;
      v8f& o = (nt == 0) ? o0 : (nt == 1) ? o1 : (nt == 2) ? o2 : o3;
      o = wmma_bf16(ap, bv, o);
    }

    __syncthreads();   // all waves done reading buf before it is refilled
  }

  // ---- epilogue: out[b, row, :] = o / l ----
#pragma unroll
  for (int i = 0; i < 8; ++i) {
    const int row = q0 + 8 * half + i;
    const float inv = 1.0f / l[i];
    float* po = out + ((size_t)(b * S_ + row)) * DOUT;
    po[l16]      = o0[i] * inv;
    po[16 + l16] = o1[i] * inv;
    po[32 + l16] = o2[i] * inv;
    po[48 + l16] = o3[i] * inv;
  }
}

// ---------------------------------------------------------------------------
extern "C" void kernel_launch(void* const* d_in, const int* in_sizes, int n_in,
                              void* d_out, int out_size, void* d_ws, size_t ws_size,
                              hipStream_t stream) {
  const float* x  = (const float*)d_in[0];
  const float* Wq = (const float*)d_in[1];
  const float* Wk = (const float*)d_in[2];
  const float* Wv = (const float*)d_in[3];
  float* out = (float*)d_out;

  // workspace: qb (2MB) | kb (2MB) | vt (2MB), all bf16
  bf16_t* qb = (bf16_t*)d_ws;
  bf16_t* kb = qb + (size_t)ROWS * DOUT;
  bf16_t* vt = kb + (size_t)ROWS * DOUT;

  // (ROWS/16) row tiles * 3 matrices * 4 col tiles = 12288 wave tasks, 8 waves/block
  qkv_kernel<<<12288 / 8, 256, 0, stream>>>(x, Wq, Wk, Wv, qb, kb, vt);
  // 1024 query tiles, 4 waves/block
  attn_kernel<<<1024 / 4, 128, 0, stream>>>(qb, kb, vt, out);
}
